// HardOC_Module_46385646797213
// MI455X (gfx1250) — matrix operand order repaired and checked by reading, not compiled, not run on hardware
//
#include <hip/hip_runtime.h>
#include <hip/hip_bf16.h>

// ---------------------------------------------------------------------------
// Problem constants
// ---------------------------------------------------------------------------
#define NB   4          // batch
#define C    512        // feat channels (= GEMM K)
#define OC   512        // out channels (= GEMM M)
#define KCLS 19         // classes
#define H    128
#define W128 128
#define HW   (H * W128)           // 16384
#define HP   130
#define HWP  (HP * HP)            // 16900 (padded output plane)
#define BN_EPS      1e-5f
#define LEAKY_SLOPE 0.01f

// GEMM tiling
#define MT   128        // M tile per block (8 waves x 16 rows)
#define NT   64         // N (pixel) tile per block (4 sub-tiles of 16)
#define KT   32         // K step (bf16 wmma depth)

typedef __attribute__((ext_vector_type(16))) __bf16 v16bf;
typedef __attribute__((ext_vector_type(8)))  float  v8f;
typedef __attribute__((ext_vector_type(4)))  unsigned int v4u;
typedef __attribute__((ext_vector_type(8)))  int    v8i;
typedef __attribute__((ext_vector_type(4)))  int    v4i;

union Frag { v16bf v; uint4 u[2]; };

// ws layout (byte offsets, all 16B aligned)
#define WS_IDX     0                       // int32 [NB*HW]              (256 KB)
#define WS_CNT     262144                  // int32 [32]
#define WS_SUMS    262272                  // float [KCLS*NB*C]          (152 KB)
#define WS_CTX     417920                  // float [KCLS*NB*OC]         (152 KB)
#define WS_STATS   573568                  // float [1024] (sum, sumsq)
#define WS_SCALE   577664                  // float [1024] (scale, shift)
#define WS_W1HI    581760                  // bf16  [OC*C]               (512 KB)
#define WS_W1LO    1106048                 // bf16  [OC*C]               (512 KB)

__device__ __forceinline__ unsigned short f32_to_bf16_rne(float x) {
    unsigned int u = __float_as_uint(x);
    u += 0x7FFFu + ((u >> 16) & 1u);           // round to nearest even
    return (unsigned short)(u >> 16);
}
__device__ __forceinline__ float bf16_bits_to_f32(unsigned short s) {
    return __uint_as_float(((unsigned int)s) << 16);
}

// ---------------------------------------------------------------------------
// Tensor Data Mover: 2D bf16 tile (tile_d0 x tile_d1 elems) -> LDS
//   D# per CDNA5 ISA §8.3/8.4 (group0: count/lds/global/type, group1: dims)
//   6-arg builtin form: (v4u g0, v8i g1, v4i g2, v4i g3, v8i g4, i32 cpol)
// ---------------------------------------------------------------------------
__device__ __forceinline__ void tdm_load_2d_bf16(unsigned lds_off, const void* gptr,
                                                 int tile_d0, int tile_d1,
                                                 int tensor_d0, int tensor_d1,
                                                 int stride_elems) {
    unsigned long long ga = (unsigned long long)(size_t)gptr;
    v4u g0;
    g0[0] = 1u;                                        // count=1, user mode
    g0[1] = lds_off;                                   // lds_addr [63:32]
    g0[2] = (unsigned)ga;                              // global_addr [95:64]
    g0[3] = (unsigned)((ga >> 32) & 0x01FFFFFFull)     // global_addr [120:96]
            | 0x80000000u;                             // type=2 ("image") [127:126]
    v8i g1;
    g1[0] = (int)(1u << 16);                           // data_size=1 (2 bytes)
    g1[1] = (int)((unsigned)(tensor_d0 & 0xFFFF) << 16);          // tensor_dim0[15:0]
    g1[2] = (int)(((unsigned)tensor_d0 >> 16) |
                  ((unsigned)(tensor_d1 & 0xFFFF) << 16));        // dim0 hi | dim1 lo
    g1[3] = (int)(((unsigned)tensor_d1 >> 16) |
                  ((unsigned)tile_d0 << 16));                     // dim1 hi | tile_dim0
    g1[4] = (int)((unsigned)tile_d1 & 0xFFFFu);        // tile_dim1 (tile_dim2=0)
    g1[5] = stride_elems;                              // tensor_dim0_stride[31:0]
    g1[6] = 0;
    g1[7] = 0;
    v4i gz4 = {0, 0, 0, 0};                            // groups 2/3: dims unused
    v8i gz8 = {0, 0, 0, 0, 0, 0, 0, 0};
    __builtin_amdgcn_tensor_load_to_lds(g0, g1, gz4, gz4, gz8, 0);
}

// ---------------------------------------------------------------------------
// 0. zero a float region
// ---------------------------------------------------------------------------
__global__ void k_zero(float* __restrict__ p, int n) {
    int i = blockIdx.x * blockDim.x + threadIdx.x;
    if (i < n) p[i] = 0.0f;
}

// ---------------------------------------------------------------------------
// 0b. zero the 1-px border ring of every output plane
// ---------------------------------------------------------------------------
__global__ void k_zero_border(float* __restrict__ y) {
    float* plane = y + (size_t)blockIdx.x * HWP;
    for (int t = threadIdx.x; t < 2 * HP + 2 * (HP - 2); t += blockDim.x) {
        int r, c;
        if (t < HP)               { r = 0;        c = t; }
        else if (t < 2 * HP)      { r = HP - 1;   c = t - HP; }
        else { int s = t - 2 * HP; r = 1 + (s >> 1); c = (s & 1) ? (HP - 1) : 0; }
        plane[r * HP + c] = 0.0f;
    }
}

// ---------------------------------------------------------------------------
// 1. per-pixel argmax over K classes + class counts
// ---------------------------------------------------------------------------
__global__ void k_argmax(const float* __restrict__ probs,
                         int* __restrict__ idx, int* __restrict__ counts) {
    int gid = blockIdx.x * blockDim.x + threadIdx.x;   // n*HW + p
    int n = gid >> 14, p = gid & (HW - 1);
    const float* base = probs + (size_t)n * KCLS * HW + p;
    float best = base[0];
    int bk = 0;
    #pragma unroll
    for (int k = 1; k < KCLS; ++k) {
        float v = base[(size_t)k * HW];
        if (v > best) { best = v; bk = k; }
    }
    idx[gid] = bk;
    atomicAdd(&counts[bk], 1);
}

// ---------------------------------------------------------------------------
// 2. sums[k][n][c] — wave-per-channel, LDS class bins
// ---------------------------------------------------------------------------
__global__ void k_masked_sums(const float* __restrict__ feats,
                              const int* __restrict__ idx,
                              float* __restrict__ sums) {
    __shared__ float bins[8][20];
    int wv = threadIdx.x >> 5, lane = threadIdx.x & 31;
    int n = blockIdx.y;
    int c = blockIdx.x * 8 + wv;
    if (lane < KCLS) bins[wv][lane] = 0.0f;   // wave-coherent init
    const float* f = feats + ((size_t)(n * C + c)) * HW;
    const int*   ix = idx + (size_t)n * HW;
    for (int p = lane; p < HW; p += 32) {
        atomicAdd(&bins[wv][ix[p]], f[p]);
    }
    if (lane < KCLS) atomicAdd(&sums[((lane * NB) + n) * C + c], bins[wv][lane]);
}

// ---------------------------------------------------------------------------
// 3. split W1 (feats half of W) into bf16 hi/lo planes
// ---------------------------------------------------------------------------
__global__ void k_wsplit(const float* __restrict__ Wc,
                         unsigned short* __restrict__ w1hi,
                         unsigned short* __restrict__ w1lo) {
    int e = blockIdx.x * blockDim.x + threadIdx.x;     // o*C + c
    int o = e >> 9, c = e & (C - 1);
    float x = Wc[(size_t)o * (2 * C) + c];
    unsigned short hi = f32_to_bf16_rne(x);
    float lof = x - bf16_bits_to_f32(hi);
    w1hi[e] = hi;
    w1lo[e] = f32_to_bf16_rne(lof);
}

// ---------------------------------------------------------------------------
// 4. ctxproj[k][n][o] = (W2[o,:] . sums[k,n,:]) / counts[k]
// ---------------------------------------------------------------------------
__global__ void k_ctxproj(const float* __restrict__ Wc,
                          const float* __restrict__ sums,
                          const int* __restrict__ counts,
                          float* __restrict__ ctx) {
    __shared__ float srow[C];
    int k = blockIdx.x, n = blockIdx.y;
    const float* sp = sums + (size_t)(k * NB + n) * C;
    for (int t = threadIdx.x; t < C; t += blockDim.x) srow[t] = sp[t];
    __syncthreads();
    int cnt = counts[k];
    float inv = 1.0f / (float)(cnt > 0 ? cnt : 1);
    for (int o = threadIdx.x; o < OC; o += blockDim.x) {
        const float* wr = Wc + (size_t)o * (2 * C) + C;   // W2 part
        float acc = 0.0f;
        #pragma unroll 4
        for (int c = 0; c < C; ++c) acc += wr[c] * srow[c];
        ctx[(size_t)(k * NB + n) * OC + o] = acc * inv;
    }
}

// ---------------------------------------------------------------------------
// 5. Main GEMM:  y_int = W1 @ feats  (bf16 hi/lo split, f32 accumulate)
//    A tiles DMA'd to LDS by the Tensor Data Mover; B tiles converted in
//    registers and pre-swizzled into fragment layout in LDS.
//    block: 256 thr (8 waves). Tile: MT=128 x NT=64, K-steps of 32.
// ---------------------------------------------------------------------------
__global__ void __launch_bounds__(256)
k_gemm(const float* __restrict__ feats,
       const unsigned short* __restrict__ w1hi,
       const unsigned short* __restrict__ w1lo,
       const float* __restrict__ ctx,
       const int* __restrict__ idx,
       float* __restrict__ y) {
    // LDS: B fragments pre-swizzled: [4 Ntile][2 plane][32 lane][16 bf16]
    __shared__ __align__(16) unsigned short ldsB[4 * 2 * 32 * 16];       // 8 KB
    // LDS: A tile, row-major [plane][128 rows][32 k]                    16 KB
    __shared__ __align__(16) unsigned short ldsA[2][MT * KT];

    const int wv   = threadIdx.x >> 5;
    const int lane = threadIdx.x & 31;
    const int half = lane >> 4;
    const int l16  = lane & 15;

    const int pixbase = blockIdx.x * NT;
    const int obase   = blockIdx.y * MT;
    const int nb      = blockIdx.z;

    v8f acc[4];
    #pragma unroll
    for (int j = 0; j < 4; ++j)
        #pragma unroll
        for (int r = 0; r < 8; ++r) acc[j][r] = 0.0f;

    const float* fbase = feats + ((size_t)nb * C) * HW + pixbase;

    for (int ks = 0; ks < C; ks += KT) {
        // ---- TDM: DMA this k-step's A tile (128 x 32 bf16, hi+lo) into LDS
        if (wv == 0) {
            tdm_load_2d_bf16((unsigned)(size_t)&ldsA[0][0],
                             w1hi + (size_t)obase * C + ks,
                             KT, MT, C, OC, C);
            tdm_load_2d_bf16((unsigned)(size_t)&ldsA[1][0],
                             w1lo + (size_t)obase * C + ks,
                             KT, MT, C, OC, C);
        }

        // ---- stage B tile (KT x NT f32) -> bf16 hi/lo, swizzled into frag layout
        #pragma unroll
        for (int it = 0; it < 2; ++it) {
            int q    = threadIdx.x + it * 256;           // 0..511
            int krow = q >> 4;                           // 0..31 local k
            int c4   = (q & 15) * 4;                     // local pixel
            const float4 f4 = *reinterpret_cast<const float4*>(
                fbase + ((size_t)(ks + krow)) * HW + c4);
            float v[4] = {f4.x, f4.y, f4.z, f4.w};
            int l   = (((krow >> 3) & 1) << 4);
            int el  = (krow & 7) | ((krow >> 4) << 3);
            #pragma unroll
            for (int e = 0; e < 4; ++e) {
                int pp = c4 + e;
                int j  = pp >> 4;
                int ll = l | (pp & 15);
                unsigned short hi = f32_to_bf16_rne(v[e]);
                float lof = v[e] - bf16_bits_to_f32(hi);
                ldsB[((j * 2 + 0) * 32 + ll) * 16 + el] = hi;
                ldsB[((j * 2 + 1) * 32 + ll) * 16 + el] = f32_to_bf16_rne(lof);
            }
        }
        // prefetch next K-slab of feats into cache (global_prefetch_b8)
        if (ks + KT < C) {
            __builtin_prefetch(fbase + ((size_t)(ks + KT + (threadIdx.x >> 4))) * HW
                                     + (threadIdx.x & 15) * 4, 0, 3);
        }
        if (wv == 0) {
            __builtin_amdgcn_s_wait_tensorcnt(0);        // A tile landed in LDS
        }
        __syncthreads();

        // ---- A fragments (hi/lo) from the DMA'd LDS tile
        Frag ahi, alo;
        {
            int arow = wv * 16 + l16;                    // local A row
            const unsigned short* r0 = &ldsA[0][arow * KT + 8 * half];
            ahi.u[0] = *reinterpret_cast<const uint4*>(r0);
            ahi.u[1] = *reinterpret_cast<const uint4*>(r0 + 16);
            const unsigned short* r1 = &ldsA[1][arow * KT + 8 * half];
            alo.u[0] = *reinterpret_cast<const uint4*>(r1);
            alo.u[1] = *reinterpret_cast<const uint4*>(r1 + 16);
        }

        // ---- 4 N-subtiles x (hi*hi + hi*lo + lo*hi)
        #pragma unroll
        for (int j = 0; j < 4; ++j) {
            Frag bhi, blo;
            const uint4* bh = reinterpret_cast<const uint4*>(&ldsB[((j * 2 + 0) * 32 + lane) * 16]);
            const uint4* bl = reinterpret_cast<const uint4*>(&ldsB[((j * 2 + 1) * 32 + lane) * 16]);
            bhi.u[0] = bh[0]; bhi.u[1] = bh[1];
            blo.u[0] = bl[0]; blo.u[1] = bl[1];
            acc[j] = __builtin_amdgcn_wmma_f32_16x16x32_bf16(
                false, ahi.v, false, bhi.v, (short)0, acc[j], false, false);
            acc[j] = __builtin_amdgcn_wmma_f32_16x16x32_bf16(
                false, ahi.v, false, blo.v, (short)0, acc[j], false, false);
            acc[j] = __builtin_amdgcn_wmma_f32_16x16x32_bf16(
                false, alo.v, false, bhi.v, (short)0, acc[j], false, false);
        }
        __syncthreads();
    }

    // ---- epilogue: + ctxproj[k*(pixel), nb, o], store into padded interior
    const int ob8 = obase + wv * 16 + 8 * half;          // first of 8 output rows
    #pragma unroll
    for (int j = 0; j < 4; ++j) {
        int pix  = pixbase + j * 16 + l16;
        int kcls = idx[nb * HW + pix];
        const float* cp = ctx + (size_t)(kcls * NB + nb) * OC + ob8;
        int h  = pix >> 7, wc = pix & 127;
        float* outp = y + ((size_t)(nb * OC + ob8)) * HWP + (h + 1) * HP + (wc + 1);
        #pragma unroll
        for (int r = 0; r < 8; ++r)
            outp[(size_t)r * HWP] = acc[j][r] + cp[r];
    }
}

// ---------------------------------------------------------------------------
// 6. BN stats over interior (ring is exactly zero -> contributes nothing)
// ---------------------------------------------------------------------------
__global__ void k_bnstats(const float* __restrict__ y, float* __restrict__ stats) {
    __shared__ float s1[256], s2[256];
    int o = blockIdx.x, nb = blockIdx.y;
    const float* base = y + ((size_t)(nb * OC + o)) * HWP + HP + 1;
    float a = 0.0f, b = 0.0f;
    for (int p = threadIdx.x; p < HW; p += 256) {
        float v = base[(p >> 7) * HP + (p & 127)];
        a += v; b += v * v;
    }
    s1[threadIdx.x] = a; s2[threadIdx.x] = b;
    __syncthreads();
    for (int st = 128; st > 0; st >>= 1) {
        if (threadIdx.x < st) {
            s1[threadIdx.x] += s1[threadIdx.x + st];
            s2[threadIdx.x] += s2[threadIdx.x + st];
        }
        __syncthreads();
    }
    if (threadIdx.x == 0) {
        atomicAdd(&stats[o], s1[0]);
        atomicAdd(&stats[OC + o], s2[0]);
    }
}

// ---------------------------------------------------------------------------
// 7. finalize: scale/shift per channel
// ---------------------------------------------------------------------------
__global__ void k_bnfinal(const float* __restrict__ stats,
                          const float* __restrict__ gamma,
                          const float* __restrict__ beta,
                          float* __restrict__ ss) {
    int o = blockIdx.x * blockDim.x + threadIdx.x;
    if (o >= OC) return;
    const float cnt = (float)(NB * HWP);
    float mu  = stats[o] / cnt;
    float var = stats[OC + o] / cnt - mu * mu;
    float rs  = rsqrtf(var + BN_EPS);
    float sc  = rs * gamma[o];
    ss[o]      = sc;
    ss[OC + o] = beta[o] - mu * sc;
}

// ---------------------------------------------------------------------------
// 8. normalize + leaky relu over the FULL padded tensor (incl. ring)
//    one block per (n,o) plane -> channel is blockIdx.x & 511
// ---------------------------------------------------------------------------
__global__ void k_norm(float* __restrict__ y, const float* __restrict__ ss) {
    int o = blockIdx.x & (OC - 1);
    float sc = ss[o], sh = ss[OC + o];
    float* plane = y + (size_t)blockIdx.x * HWP;
    for (int t = threadIdx.x; t < HWP; t += blockDim.x) {
        float v = plane[t] * sc + sh;
        plane[t] = v > 0.0f ? v : LEAKY_SLOPE * v;
    }
}

// ---------------------------------------------------------------------------
extern "C" void kernel_launch(void* const* d_in, const int* in_sizes, int n_in,
                              void* d_out, int out_size, void* d_ws, size_t ws_size,
                              hipStream_t stream) {
    const float* feats = (const float*)d_in[0];
    const float* probs = (const float*)d_in[1];
    const float* Wc    = (const float*)d_in[2];
    const float* gamma = (const float*)d_in[3];
    const float* beta  = (const float*)d_in[4];
    float* y = (float*)d_out;

    char* ws = (char*)d_ws;
    int*            idx   = (int*)(ws + WS_IDX);
    int*            cnts  = (int*)(ws + WS_CNT);
    float*          sums  = (float*)(ws + WS_SUMS);
    float*          ctx   = (float*)(ws + WS_CTX);
    float*          stats = (float*)(ws + WS_STATS);
    float*          ss    = (float*)(ws + WS_SCALE);
    unsigned short* w1hi  = (unsigned short*)(ws + WS_W1HI);
    unsigned short* w1lo  = (unsigned short*)(ws + WS_W1LO);

    // 0. zero counts+sums (contiguous) and stats; zero output border ring
    {
        int nz = 32 + KCLS * NB * C;
        k_zero<<<(nz + 255) / 256, 256, 0, stream>>>((float*)(ws + WS_CNT), nz);
        k_zero<<<4, 256, 0, stream>>>(stats, 1024);
        k_zero_border<<<NB * OC, 256, 0, stream>>>(y);
    }
    // 1. per-pixel argmax + counts
    k_argmax<<<(NB * HW) / 256, 256, 0, stream>>>(probs, idx, cnts);
    // 2. masked channel sums
    k_masked_sums<<<dim3(C / 8, NB), 256, 0, stream>>>(feats, idx, sums);
    // 3. bf16 hi/lo weight planes
    k_wsplit<<<(OC * C) / 256, 256, 0, stream>>>(Wc, w1hi, w1lo);
    // 4. context projection table (tiny GEMM)
    k_ctxproj<<<dim3(KCLS, NB), 256, 0, stream>>>(Wc, sums, cnts, ctx);
    // 5. main WMMA GEMM (TDM-fed A tiles) + ctx add -> padded interior of d_out
    k_gemm<<<dim3(HW / NT, OC / MT, NB), 256, 0, stream>>>(feats, w1hi, w1lo, ctx, idx, y);
    // 6-8. batchnorm + leaky relu
    k_bnstats<<<dim3(OC, NB), 256, 0, stream>>>(y, stats);
    k_bnfinal<<<2, 256, 0, stream>>>(stats, gamma, beta, ss);
    k_norm<<<NB * OC, 256, 0, stream>>>(y, ss);
}
